// TransformerBlock_5299989643595
// MI455X (gfx1250) — matrix operand order repaired and checked by reading, not compiled, not run on hardware
//
#include <hip/hip_runtime.h>
#include <math.h>

// ---------------------------------------------------------------------------
// Problem constants (match reference)
// ---------------------------------------------------------------------------
constexpr int cS   = 2048;
constexpr int cD   = 2048;
constexpr int cH   = 16;
constexpr int cHD  = 128;
constexpr int cE   = 64;
constexpr int cK   = 2;
constexpr int cHID = 512;
constexpr int cCAP = 256;
constexpr int cT   = cS;          // tokens (B=1)
constexpr float cEPS = 1e-6f;

// ---------------------------------------------------------------------------
// CDNA5 WMMA types
// ---------------------------------------------------------------------------
typedef __attribute__((ext_vector_type(16))) _Float16 v16h;
typedef __attribute__((ext_vector_type(8)))  _Float16 v8h;
typedef __attribute__((ext_vector_type(8)))  float    v8f;

__device__ inline v8h cvt8(float4 a, float4 b, float s) {
  v8h r;
  r[0] = (_Float16)(a.x * s); r[1] = (_Float16)(a.y * s);
  r[2] = (_Float16)(a.z * s); r[3] = (_Float16)(a.w * s);
  r[4] = (_Float16)(b.x * s); r[5] = (_Float16)(b.y * s);
  r[6] = (_Float16)(b.z * s); r[7] = (_Float16)(b.w * s);
  return r;
}

// ---------------------------------------------------------------------------
// General WMMA GEMM:  C[M,N] (f32) = A[M,K] (f32,row-major) * B[N,K]^T
//   - f32 operands converted to f16 while staging to LDS; f32 accumulate
//   - optional residual add (Cres), optional gather-A (gIdx/gScale per batch),
//     optional scatter-add epilogue (sTok -> atomicAdd rows into C)
//   - blockIdx.z batches (experts); Kdim must be a multiple of 64
// Tile: 128x128, K-tile 64 (2 x K=32 WMMA sub-steps per staged tile).
// 256 threads = 8 wave32 waves, each wave owns a 32x64 patch = 2x4 grid of
// 16x16 accumulators -> 16 v_wmma_f32_16x16x32_f16 per K-tile per wave.
// OOB rows are zero-filled in LDS once before the K-loop (loop-invariant).
// ---------------------------------------------------------------------------
#define LDSK 72   // 64 + 8 halves pad: row stride 144B (16B aligned, bank-skewed)

__global__ __launch_bounds__(256)
void gemm_wmma(const float* __restrict__ A, long lda, long aBatch,
               const float* __restrict__ B, long ldb, long bBatch,
               float* __restrict__ C, long ldc, long cBatch,
               int M, int N, int Kdim,
               const float* __restrict__ Cres,
               const int*   __restrict__ gIdx,
               const float* __restrict__ gScale,
               const int*   __restrict__ sTok)
{
  __shared__ _Float16 As[128 * LDSK];
  __shared__ _Float16 Bs[128 * LDSK];

  const int z    = blockIdx.z;
  const int bm   = blockIdx.y * 128;
  const int bn   = blockIdx.x * 128;
  const int tid  = threadIdx.x;
  const int lane = tid & 31;            // wave32
  const int wave = tid >> 5;            // 8 waves
  const int wm   = (wave & 3) * 32;     // wave row offset in tile
  const int wn   = (wave >> 2) * 64;    // wave col offset in tile

  // staging coords: 2 threads per tile row, 32 f32 each
  const int srow = tid >> 1;            // 0..127
  const int scol = (tid & 1) * 32;      // 0 or 32

  // A source row (optionally gathered + scaled)
  const long aRow = bm + srow;
  const float* aPtr = A;
  float aScl = 1.0f;
  const bool aOK = (aRow < (long)M);
  if (aOK) {
    if (gIdx) {
      const int tok = gIdx[(long)z * M + aRow];
      aPtr = A + (long)tok * lda;
      aScl = gScale[(long)z * M + aRow];
    } else {
      aPtr = A + (long)z * aBatch + aRow * lda;
    }
  }
  const long bRow = bn + srow;
  const bool bOK = (bRow < (long)N);
  const float* bPtr = bOK ? (B + (long)z * bBatch + bRow * ldb) : B;

  _Float16* aDst = As + srow * LDSK + scol;
  _Float16* bDst = Bs + srow * LDSK + scol;

  // zero-fill OOB rows ONCE (loop body never overwrites them)
  if (!aOK) {
    #pragma unroll
    for (int e = 0; e < 32; ++e) aDst[e] = (_Float16)0.0f;
  }
  if (!bOK) {
    #pragma unroll
    for (int e = 0; e < 32; ++e) bDst[e] = (_Float16)0.0f;
  }

  // hoisted LDS fragment base pointers (per ISA A/B lane layouts)
  // A frag (16x32): lanes 0-15 rows M, K{0..7,16..23}; lanes 16-31 K{8..15,24..31}
  const _Float16* aFrag0 = As + (wm + (lane & 15)) * LDSK + ((lane >> 4) << 3);
  const _Float16* aFrag1 = aFrag0 + 16 * LDSK;
  // B frag (32x16): lane&15 selects column N; lanes 0-15 K{0..7,16..23}; lanes 16-31 K{8..15,24..31}
  const _Float16* bFrag[4];
  #pragma unroll
  for (int ni = 0; ni < 4; ++ni)
    bFrag[ni] = Bs + (wn + ni * 16 + (lane & 15)) * LDSK + ((lane >> 4) << 3);

  v8f acc[2][4];
  #pragma unroll
  for (int mi = 0; mi < 2; ++mi)
    #pragma unroll
    for (int ni = 0; ni < 4; ++ni)
      #pragma unroll
      for (int e = 0; e < 8; ++e) acc[mi][ni][e] = 0.0f;

  for (int kt = 0; kt < Kdim; kt += 64) {
    // ---- stage A tile (f32 -> f16, scaled) ----
    if (aOK) {
      const float4* gp = (const float4*)(aPtr + kt + scol);
      float4 f0 = gp[0], f1 = gp[1], f2 = gp[2], f3 = gp[3];
      float4 f4 = gp[4], f5 = gp[5], f6 = gp[6], f7 = gp[7];
      if (kt + 128 < Kdim) __builtin_prefetch(aPtr + kt + 128, 0, 0);
      *(v8h*)(aDst)      = cvt8(f0, f1, aScl);
      *(v8h*)(aDst + 8)  = cvt8(f2, f3, aScl);
      *(v8h*)(aDst + 16) = cvt8(f4, f5, aScl);
      *(v8h*)(aDst + 24) = cvt8(f6, f7, aScl);
    }
    // ---- stage B tile (f32 -> f16) ----
    if (bOK) {
      const float4* gp = (const float4*)(bPtr + kt + scol);
      float4 f0 = gp[0], f1 = gp[1], f2 = gp[2], f3 = gp[3];
      float4 f4 = gp[4], f5 = gp[5], f6 = gp[6], f7 = gp[7];
      if (kt + 128 < Kdim) __builtin_prefetch(bPtr + kt + 128, 0, 0);
      *(v8h*)(bDst)      = cvt8(f0, f1, 1.0f);
      *(v8h*)(bDst + 8)  = cvt8(f2, f3, 1.0f);
      *(v8h*)(bDst + 16) = cvt8(f4, f5, 1.0f);
      *(v8h*)(bDst + 24) = cvt8(f6, f7, 1.0f);
    }
    __syncthreads();

    // ---- 2 x (K=32) WMMA sub-steps: 16 wmma per K-tile per wave ----
    #pragma unroll
    for (int ks = 0; ks < 2; ++ks) {
      const int kofs = ks * 32;
      v16h af[2];
      {
        const _Float16* p0 = aFrag0 + kofs;
        v8h lo = *(const v8h*)p0;
        v8h hi = *(const v8h*)(p0 + 16);
        v16h a;
        #pragma unroll
        for (int e = 0; e < 8; ++e) { a[e] = lo[e]; a[8 + e] = hi[e]; }
        af[0] = a;
        const _Float16* p1 = aFrag1 + kofs;
        lo = *(const v8h*)p1;
        hi = *(const v8h*)(p1 + 16);
        #pragma unroll
        for (int e = 0; e < 8; ++e) { a[e] = lo[e]; a[8 + e] = hi[e]; }
        af[1] = a;
      }
      v16h bfr[4];
      #pragma unroll
      for (int ni = 0; ni < 4; ++ni) {
        const _Float16* p = bFrag[ni] + kofs;
        v8h b0 = *(const v8h*)p;
        v8h b1 = *(const v8h*)(p + 16);
        v16h b;
        #pragma unroll
        for (int e = 0; e < 8; ++e) { b[e] = b0[e]; b[8 + e] = b1[e]; }
        bfr[ni] = b;
      }
      #pragma unroll
      for (int mi = 0; mi < 2; ++mi)
        #pragma unroll
        for (int ni = 0; ni < 4; ++ni)
          acc[mi][ni] = __builtin_amdgcn_wmma_f32_16x16x32_f16(
              /*neg_a=*/false, af[mi], /*neg_b=*/false, bfr[ni],
              /*c_mod=*/(short)0, acc[mi][ni],
              /*reuse_a=*/false, /*reuse_b=*/false);
    }
    __syncthreads();
  }

  // ---- epilogue: C layout (16x16 f32): VGPR v -> M = v + 8*(lane>=16),
  //      N = lane&15 ----
  const int rbase = (lane >> 4) * 8;
  const int cl    = lane & 15;
  #pragma unroll
  for (int mi = 0; mi < 2; ++mi) {
    #pragma unroll
    for (int ni = 0; ni < 4; ++ni) {
      const int gm0 = bm + wm + mi * 16 + rbase;
      const int gn  = bn + wn + ni * 16 + cl;
      #pragma unroll
      for (int v = 0; v < 8; ++v) {
        const int gm = gm0 + v;
        if (gm < M && gn < N) {
          float val = acc[mi][ni][v];
          if (Cres) val += Cres[(long)z * cBatch + (long)gm * ldc + gn];
          if (sTok) {
            const int tok = sTok[(long)z * M + gm];
            atomicAdd(C + (long)tok * ldc + gn, val);  // global_atomic_add_f32
          } else {
            C[(long)z * cBatch + (long)gm * ldc + gn] = val;
          }
        }
      }
    }
  }
}

// ---------------------------------------------------------------------------
// RMSNorm: one block per row
// ---------------------------------------------------------------------------
__global__ __launch_bounds__(256)
void rmsnorm_kernel(const float* __restrict__ x, const float* __restrict__ w,
                    float* __restrict__ o, int D)
{
  const int row = blockIdx.x;
  const float* xr = x + (long)row * D;
  float* orow = o + (long)row * D;
  __shared__ float red[256];
  float s = 0.0f;
  for (int c = threadIdx.x; c < D; c += 256) { float v = xr[c]; s += v * v; }
  red[threadIdx.x] = s; __syncthreads();
  for (int off = 128; off > 0; off >>= 1) {
    if (threadIdx.x < off) red[threadIdx.x] += red[threadIdx.x + off];
    __syncthreads();
  }
  const float rs = rsqrtf(red[0] / (float)D + cEPS);
  for (int c = threadIdx.x; c < D; c += 256) orow[c] = xr[c] * rs * w[c];
}

// ---------------------------------------------------------------------------
// RoPE in place on (S, H, HD); thread per rotation pair
// ---------------------------------------------------------------------------
__global__ __launch_bounds__(256)
void rope_kernel(float* __restrict__ t,
                 const float* __restrict__ fc, const float* __restrict__ fs)
{
  const long n = (long)cS * cH * (cHD / 2);
  const long i = (long)blockIdx.x * blockDim.x + threadIdx.x;
  if (i >= n) return;
  const int  p  = (int)(i % (cHD / 2));
  const long sh = i / (cHD / 2);
  const int  hh = (int)(sh % cH);
  const long s  = sh / cH;
  const float c  = fc[s * (cHD / 2) + p];
  const float sn = fs[s * (cHD / 2) + p];
  float* base = t + s * (long)(cH * cHD) + hh * cHD;
  const float x0 = base[2 * p], x1 = base[2 * p + 1];
  base[2 * p]     = x0 * c - x1 * sn;
  base[2 * p + 1] = x0 * sn + x1 * c;
}

// ---------------------------------------------------------------------------
// V transpose per head: vt[d][s] = v[s, head, d]
// ---------------------------------------------------------------------------
__global__ __launch_bounds__(256)
void transpose_v_kernel(const float* __restrict__ v, float* __restrict__ vt,
                        int head)
{
  const long n = (long)cHD * cS;
  const long i = (long)blockIdx.x * blockDim.x + threadIdx.x;
  if (i >= n) return;
  const int  d = (int)(i % cHD);
  const long s = i / cHD;
  vt[(long)d * cS + s] = v[s * (long)cD + head * cHD + d];
}

// ---------------------------------------------------------------------------
// Causal softmax in place, one block per row; logit scale folded in
// ---------------------------------------------------------------------------
__global__ __launch_bounds__(256)
void softmax_causal_kernel(float* __restrict__ sc, float scale)
{
  const int r = blockIdx.x;
  float* row = sc + (long)r * cS;
  __shared__ float red[256];
  const int tid = threadIdx.x;

  float m = -3.0e38f;
  for (int c = tid; c < cS; c += 256) {
    const float v = (c <= r) ? row[c] * scale : -3.0e38f;
    m = fmaxf(m, v);
  }
  red[tid] = m; __syncthreads();
  for (int off = 128; off > 0; off >>= 1) {
    if (tid < off) red[tid] = fmaxf(red[tid], red[tid + off]);
    __syncthreads();
  }
  m = red[0]; __syncthreads();

  float sum = 0.0f;
  for (int c = tid; c < cS; c += 256) {
    if (c <= r) { const float e = __expf(row[c] * scale - m); row[c] = e; sum += e; }
    else row[c] = 0.0f;
  }
  red[tid] = sum; __syncthreads();
  for (int off = 128; off > 0; off >>= 1) {
    if (tid < off) red[tid] += red[tid + off];
    __syncthreads();
  }
  const float inv = 1.0f / red[0];
  for (int c = tid; c <= r; c += 256) row[c] *= inv;
}

// ---------------------------------------------------------------------------
// Router: softmax over E=64 logits + top-2 (stable ties like jax top_k)
// ---------------------------------------------------------------------------
__global__ __launch_bounds__(64)
void router_topk_kernel(const float* __restrict__ logits,
                        int* __restrict__ top_idx, float* __restrict__ top_w)
{
  const int t = blockIdx.x;
  __shared__ float p[cE];
  const int e = threadIdx.x;
  p[e] = logits[(long)t * cE + e];
  __syncthreads();
  if (e == 0) {
    float m = -3.0e38f;
    for (int i = 0; i < cE; ++i) m = fmaxf(m, p[i]);
    float s = 0.0f;
    for (int i = 0; i < cE; ++i) { p[i] = __expf(p[i] - m); s += p[i]; }
    const float inv = 1.0f / s;
    int i0 = 0, i1 = 0; float v0 = -1.0f, v1 = -1.0f;
    for (int i = 0; i < cE; ++i) {
      const float pv = p[i] * inv;
      if (pv > v0)      { v1 = v0; i1 = i0; v0 = pv; i0 = i; }
      else if (pv > v1) { v1 = pv; i1 = i; }
    }
    top_idx[t * 2] = i0; top_idx[t * 2 + 1] = i1;
    top_w[t * 2] = v0;   top_w[t * 2 + 1] = v1;
  }
}

// ---------------------------------------------------------------------------
// Capacity assignment: one thread per expert scans assignments in flat
// (token-major) order == reference's stable argsort-by-expert. Pads with
// token 0 / weight 0 (zero rows -> zero expert output, like the reference).
// ---------------------------------------------------------------------------
__global__ __launch_bounds__(64)
void assign_slots_kernel(const int* __restrict__ top_idx,
                         const float* __restrict__ top_w,
                         int* __restrict__ slot_tok, float* __restrict__ slot_w)
{
  const int e = threadIdx.x;
  int cnt = 0;
  for (int i = 0; i < cT * cK; ++i) {
    if (top_idx[i] == e) {
      if (cnt < cCAP) {
        slot_tok[e * cCAP + cnt] = i / cK;
        slot_w[e * cCAP + cnt]   = top_w[i];
      }
      ++cnt;
    }
  }
  for (int c = cnt; c < cCAP; ++c) {
    slot_tok[e * cCAP + c] = 0;
    slot_w[e * cCAP + c]   = 0.0f;
  }
}

// ---------------------------------------------------------------------------
// g = silu(h1) * h3, written into h1
// ---------------------------------------------------------------------------
__global__ __launch_bounds__(256)
void silu_gate_kernel(float* __restrict__ h1, const float* __restrict__ h3,
                      long n)
{
  const long i = (long)blockIdx.x * blockDim.x + threadIdx.x;
  if (i >= n) return;
  const float a = h1[i];
  const float s = a / (1.0f + __expf(-a));
  h1[i] = s * h3[i];
}

// ---------------------------------------------------------------------------
// out = h  (init before MoE scatter-adds)
// ---------------------------------------------------------------------------
__global__ __launch_bounds__(256)
void copy_kernel(const float* __restrict__ src, float* __restrict__ dst, long n)
{
  const long i = (long)blockIdx.x * blockDim.x + threadIdx.x;
  if (i < n) dst[i] = src[i];
}

// ---------------------------------------------------------------------------
// Host orchestration
// ---------------------------------------------------------------------------
static inline unsigned cdiv(long a, long b) { return (unsigned)((a + b - 1) / b); }

extern "C" void kernel_launch(void* const* d_in, const int* in_sizes, int n_in,
                              void* d_out, int out_size, void* d_ws, size_t ws_size,
                              hipStream_t stream)
{
  (void)in_sizes; (void)n_in; (void)out_size; (void)ws_size;
  const float* x    = (const float*)d_in[0];
  const float* fcos = (const float*)d_in[1];
  const float* fsin = (const float*)d_in[2];
  const float* anw  = (const float*)d_in[3];
  const float* fnw  = (const float*)d_in[4];
  const float* wq   = (const float*)d_in[5];
  const float* wk   = (const float*)d_in[6];
  const float* wv   = (const float*)d_in[7];
  const float* wo   = (const float*)d_in[8];
  const float* rw   = (const float*)d_in[9];
  const float* w1   = (const float*)d_in[10];
  const float* w2   = (const float*)d_in[11];
  const float* w3   = (const float*)d_in[12];
  float* out = (float*)d_out;

  // workspace carve-up (~203 MB)
  char* ws = (char*)d_ws;
  size_t off = 0;
  auto alloc = [&](size_t bytes) -> void* {
    void* p = ws + off;
    off += (bytes + 255) & ~(size_t)255;
    return p;
  };
  float* xn     = (float*)alloc((size_t)cS * cD * 4);
  float* q      = (float*)alloc((size_t)cS * cD * 4);
  float* k      = (float*)alloc((size_t)cS * cD * 4);
  float* v      = (float*)alloc((size_t)cS * cD * 4);
  float* scores = (float*)alloc((size_t)cS * cS * 4);
  float* vt     = (float*)alloc((size_t)cHD * cS * 4);
  float* attn   = (float*)alloc((size_t)cS * cD * 4);
  float* h      = (float*)alloc((size_t)cS * cD * 4);
  float* hn     = (float*)alloc((size_t)cS * cD * 4);
  float* rlog   = (float*)alloc((size_t)cT * cE * 4);
  int*   tidx   = (int*)  alloc((size_t)cT * cK * 4);
  float* tw     = (float*)alloc((size_t)cT * cK * 4);
  int*   stok   = (int*)  alloc((size_t)cE * cCAP * 4);
  float* sw     = (float*)alloc((size_t)cE * cCAP * 4);
  float* h1b    = (float*)alloc((size_t)cE * cCAP * cHID * 4);
  float* h3b    = (float*)alloc((size_t)cE * cCAP * cHID * 4);

  const dim3 blk(256);

  // 1) attn RMSNorm
  rmsnorm_kernel<<<cS, blk, 0, stream>>>(x, anw, xn, cD);

  // 2) Q/K/V projections (WMMA)
  const dim3 gQKV(cD / 128, cS / 128, 1);
  gemm_wmma<<<gQKV, blk, 0, stream>>>(xn, cD, 0, wq, cD, 0, q, cD, 0,
                                      cS, cD, cD, nullptr, nullptr, nullptr, nullptr);
  gemm_wmma<<<gQKV, blk, 0, stream>>>(xn, cD, 0, wk, cD, 0, k, cD, 0,
                                      cS, cD, cD, nullptr, nullptr, nullptr, nullptr);
  gemm_wmma<<<gQKV, blk, 0, stream>>>(xn, cD, 0, wv, cD, 0, v, cD, 0,
                                      cS, cD, cD, nullptr, nullptr, nullptr, nullptr);

  // 3) RoPE on q and k
  const long nRope = (long)cS * cH * (cHD / 2);
  rope_kernel<<<cdiv(nRope, 256), blk, 0, stream>>>(q, fcos, fsin);
  rope_kernel<<<cdiv(nRope, 256), blk, 0, stream>>>(k, fcos, fsin);

  // 4) attention, head by head (scores stay L2-resident between kernels)
  const float smScale = 0.08838834764831845f;  // 1/sqrt(128)
  for (int head = 0; head < cH; ++head) {
    transpose_v_kernel<<<cdiv((long)cHD * cS, 256), blk, 0, stream>>>(v, vt, head);
    // scores = Q_h * K_h^T  (M=S, N=S, K=HD)
    gemm_wmma<<<dim3(cS / 128, cS / 128, 1), blk, 0, stream>>>(
        q + head * cHD, cD, 0, k + head * cHD, cD, 0, scores, cS, 0,
        cS, cS, cHD, nullptr, nullptr, nullptr, nullptr);
    softmax_causal_kernel<<<cS, blk, 0, stream>>>(scores, smScale);
    // attn_h = P * V  (via vt: C = P * vt^T; M=S, N=HD, K=S)
    gemm_wmma<<<dim3(cHD / 128, cS / 128, 1), blk, 0, stream>>>(
        scores, cS, 0, vt, cS, 0, attn + head * cHD, cD, 0,
        cS, cHD, cS, nullptr, nullptr, nullptr, nullptr);
  }

  // 5) h = x + attn * Wo^T   (residual fused in epilogue)
  gemm_wmma<<<gQKV, blk, 0, stream>>>(attn, cD, 0, wo, cD, 0, h, cD, 0,
                                      cS, cD, cD, x, nullptr, nullptr, nullptr);

  // 6) ffn RMSNorm
  rmsnorm_kernel<<<cS, blk, 0, stream>>>(h, fnw, hn, cD);

  // 7) router logits (M=T, N=E, K=D)
  gemm_wmma<<<dim3(1, cS / 128, 1), blk, 0, stream>>>(
      hn, cD, 0, rw, cD, 0, rlog, cE, 0,
      cT, cE, cD, nullptr, nullptr, nullptr, nullptr);
  router_topk_kernel<<<cT, dim3(64), 0, stream>>>(rlog, tidx, tw);
  assign_slots_kernel<<<1, dim3(64), 0, stream>>>(tidx, tw, stok, sw);

  // 8) expert GEMMs, batched over z = expert; A rows gathered+scaled from hn
  const long w13Stride = (long)cHID * cD;
  const long hStride   = (long)cCAP * cHID;
  gemm_wmma<<<dim3(cHID / 128, cCAP / 128, cE), blk, 0, stream>>>(
      hn, cD, 0, w1, cD, w13Stride, h1b, cHID, hStride,
      cCAP, cHID, cD, nullptr, stok, sw, nullptr);
  gemm_wmma<<<dim3(cHID / 128, cCAP / 128, cE), blk, 0, stream>>>(
      hn, cD, 0, w3, cD, w13Stride, h3b, cHID, hStride,
      cCAP, cHID, cD, nullptr, stok, sw, nullptr);

  // 9) g = silu(h1) * h3  (into h1b)
  const long nGate = (long)cE * cCAP * cHID;
  silu_gate_kernel<<<cdiv(nGate, 256), blk, 0, stream>>>(h1b, h3b, nGate);

  // 10) out = h, then expert down-proj scatter-adds routed outputs into out
  const long nOut = (long)cS * cD;
  copy_kernel<<<cdiv(nOut, 256), blk, 0, stream>>>(h, out, nOut);
  gemm_wmma<<<dim3(cD / 128, cCAP / 128, cE), blk, 0, stream>>>(
      h1b, cHID, hStride, w2, cHID, (long)cD * cHID, out, cD, 0,
      cCAP, cD, cHID, nullptr, nullptr, nullptr, stok);
}
